// FFiNetModel_87273735455118
// MI455X (gfx1250) — compile-verified, matrix operand order checked by reading, not run on hardware
//
#include <hip/hip_runtime.h>
#include <hip/hip_bf16.h>

// =====================  types / helpers  =====================

typedef __attribute__((ext_vector_type(16))) __bf16 v16bf;
typedef __attribute__((ext_vector_type(8)))  float  v8f;

union Frag16 {            // 16 bf16 lanes-worth of one WMMA A/B fragment
    v16bf v;
    uint4 q[2];
};

#define HD 128
#define NH 8

static __device__ __forceinline__ unsigned short f2bf(float f) {
    unsigned int u = __float_as_uint(f);
    unsigned int r = u + 0x7FFFu + ((u >> 16) & 1u);   // round-to-nearest-even
    return (unsigned short)(r >> 16);
}

// ordered-uint encoding of float for atomicMax over signed floats
static __device__ __forceinline__ unsigned int encF(float f) {
    unsigned int u = __float_as_uint(f);
    return (u & 0x80000000u) ? ~u : (u | 0x80000000u);
}
static __device__ __forceinline__ float decF(unsigned int u) {
    return __uint_as_float((u & 0x80000000u) ? (u & 0x7FFFFFFFu) : ~u);
}

static __device__ __forceinline__ float leaky02(float t) {
    return t >= 0.f ? t : 0.2f * t;
}

static __device__ __forceinline__ float dist3(const float* pos, int a, int b) {
    float dx = pos[3*a+0] - pos[3*b+0];
    float dy = pos[3*a+1] - pos[3*b+1];
    float dz = pos[3*a+2] - pos[3*b+2];
    return sqrtf(dx*dx + dy*dy + dz*dz + 1e-12f);
}

// =====================  conversion kernels  =====================

// x -> bf16 (row major) ; 7 weight mats (128x128) -> bf16 TRANSPOSED (n-major)
__global__ void k_convert(const float* __restrict__ x,
                          const float* w0, const float* w1, const float* w2, const float* w3,
                          const float* w4, const float* w5, const float* w6,
                          unsigned short* __restrict__ xbf,
                          unsigned short* __restrict__ wbf, int n_x) {
    int i = blockIdx.x * blockDim.x + threadIdx.x;
    if (i < n_x) { xbf[i] = f2bf(x[i]); return; }
    int wi = i - n_x;
    if (wi >= 7 * HD * HD) return;
    int mat = wi >> 14;
    int r   = wi & 16383;
    int n = r >> 7, k = r & 127;
    const float* W = (mat == 0) ? w0 : (mat == 1) ? w1 : (mat == 2) ? w2 :
                     (mat == 3) ? w3 : (mat == 4) ? w4 : (mat == 5) ? w5 : w6;
    wbf[mat * 16384 + n * HD + k] = f2bf(W[k * HD + n]);   // transpose
}

__global__ void k_f2bf(const float* __restrict__ src, unsigned short* __restrict__ dst, int n) {
    int i = blockIdx.x * blockDim.x + threadIdx.x;
    if (i < n) dst[i] = f2bf(src[i]);
}

// =====================  WMMA projection:  P[mat] = x @ W[mat] + pe[mat]  =====================
// One wave per (mat, 16-row M-tile). A fragments (shared across all 8 N-tiles) are
// loaded once into registers and reused for 32 WMMAs -> 8x less A traffic.

__global__ void k_proj(const unsigned short* __restrict__ xbf,
                       const unsigned short* __restrict__ wbf,
                       float* __restrict__ projBase, int nRows, int mTiles) {
    int wave = (int)((blockIdx.x * blockDim.x + threadIdx.x) >> 5);
    int lane = threadIdx.x & 31;
    if (wave >= mTiles * 4) return;            // wave-uniform: EXEC all-1s for WMMA
    int mat = wave / mTiles;
    int mT  = wave - mat * mTiles;
    int m0 = mT * 16;
    int lr = lane & 15;
    int kb = (lane >> 4) << 3;                 // 0 or 8

    // load all four K-step A fragments for this 16-row slab (K = 0..127)
    Frag16 a[4];
    const unsigned short* Abase = xbf + (size_t)(m0 + lr) * HD;
#pragma unroll
    for (int ks = 0; ks < 4; ++ks) {
        a[ks].q[0] = *(const uint4*)(Abase + ks * 32 + kb);
        a[ks].q[1] = *(const uint4*)(Abase + ks * 32 + 16 + kb);
    }

    const unsigned short* Wmat = wbf + (size_t)mat * 16384;
    float* outp = projBase + (size_t)mat * nRows * HD;

#pragma unroll
    for (int nT = 0; nT < 8; ++nT) {
        int n0 = nT * 16;
        const unsigned short* Bbase = Wmat + (size_t)(n0 + lr) * HD;
        v8f acc = {};
#pragma unroll
        for (int ks = 0; ks < 4; ++ks) {
            Frag16 b;
            b.q[0] = *(const uint4*)(Bbase + ks * 32 + kb);
            b.q[1] = *(const uint4*)(Bbase + ks * 32 + 16 + kb);
            acc = __builtin_amdgcn_wmma_f32_16x16x32_bf16(false, a[ks].v, false, b.v,
                                                          (short)0, acc, false, false);
        }
        // sinusoidal positional encoding, row index == mat (order: src, mid2, mid1, dst)
        // ang is in [0, 3] so hardware fast transcendentals are exact enough here
        int n = n0 + lr;
        float div = __expf((float)(n & ~1) * (-9.210340371976184f / 128.0f));
        float ang = (float)mat * div;
        float pe  = (n & 1) ? __cosf(ang) : __sinf(ang);
#pragma unroll
        for (int v = 0; v < 8; ++v) {
            int row = m0 + v + ((lane >> 4) << 3);
            outp[(size_t)row * HD + n] = acc[v] + pe;
        }
    }
}

// =====================  hop logits kernels (wave per item, 4 channels/lane)  =====================

__global__ void k_edge_logits(const int* __restrict__ ei, const unsigned char* __restrict__ eattr,
                              const float* __restrict__ pos,
                              const float* __restrict__ mid1P, const float* __restrict__ dstP,
                              const float* __restrict__ Wpb, const float* __restrict__ bpb,
                              const float* __restrict__ Wpu, const float* __restrict__ bpu,
                              const float* __restrict__ aD,
                              float* __restrict__ logitsBuf, unsigned int* __restrict__ gmax, int E) {
    int w = (int)((blockIdx.x * blockDim.x + threadIdx.x) >> 5);
    int lane = threadIdx.x & 31;
    if (w >= E) return;
    int es = ei[w], ed = ei[E + w];
    float d1 = dist3(pos, es, ed);
    float m  = eattr[w] ? 1.f : 0.f;
    float d2 = d1 * d1;
    float p1 = 1.f / d1, p2 = p1 * p1, p6 = p2 * p2 * p2, p12 = p6 * p6;
    float acc = 0.f;
    int cb = lane * 4;
#pragma unroll
    for (int j = 0; j < 4; ++j) {
        int c = cb + j;
        float dm = (1.f - m) * (d1 * Wpb[c] + d2 * Wpb[HD + c]) + bpb[c]
                 + m * (p6 * Wpu[c] + p12 * Wpu[HD + c] + p1 * Wpu[2 * HD + c]) + bpu[c];
        float t = (mid1P[(size_t)es * HD + c] + dstP[(size_t)ed * HD + c]) * dm;
        acc += aD[c] * leaky02(t);
    }
    acc += __shfl_xor(acc, 1);
    acc += __shfl_xor(acc, 2);
    if ((lane & 3) == 0) {
        logitsBuf[(size_t)w * NH + (lane >> 2)] = acc;
        atomicMax(gmax, encF(acc));
    }
}

__global__ void k_triple_logits(const int* __restrict__ ti, const float* __restrict__ pos,
                                const float* __restrict__ mid2P, const float* __restrict__ mid1P,
                                const float* __restrict__ dstP,
                                const float* __restrict__ Wpu1, const float* __restrict__ bpu1,
                                const float* __restrict__ Wang, const float* __restrict__ bang,
                                const float* __restrict__ aT,
                                float* __restrict__ logitsBuf, unsigned int* __restrict__ gmax, int T) {
    int w = (int)((blockIdx.x * blockDim.x + threadIdx.x) >> 5);
    int lane = threadIdx.x & 31;
    if (w >= T) return;
    int ts = ti[w], tm = ti[T + w], td = ti[2 * T + w];
    float dd = dist3(pos, ts, td);
    float p1 = 1.f / dd, p2 = p1 * p1, p6 = p2 * p2 * p2, p12 = p6 * p6;
    float vax = pos[3*ts+0] - pos[3*tm+0], vay = pos[3*ts+1] - pos[3*tm+1], vaz = pos[3*ts+2] - pos[3*tm+2];
    float vbx = pos[3*td+0] - pos[3*tm+0], vby = pos[3*td+1] - pos[3*tm+1], vbz = pos[3*td+2] - pos[3*tm+2];
    float dot = vax * vbx + vay * vby + vaz * vbz;
    float na = dist3(pos, ts, tm), nb = dist3(pos, td, tm);
    float c = dot / (na * nb + 1e-9f);
    c = fminf(fmaxf(c, -1.f + 1e-6f), 1.f - 1e-6f);
    float th = acosf(c), th2 = th * th;
    float acc = 0.f;
    int cb = lane * 4;
#pragma unroll
    for (int j = 0; j < 4; ++j) {
        int cc = cb + j;
        float dm2 = p6 * Wpu1[cc] + p12 * Wpu1[HD + cc] + p1 * Wpu1[2 * HD + cc] + bpu1[cc];
        float am  = th * Wang[cc] + th2 * Wang[HD + cc] + bang[cc];
        float t = (mid2P[(size_t)ts * HD + cc] + dstP[(size_t)td * HD + cc] +
                   mid1P[(size_t)tm * HD + cc]) * (am + dm2);
        acc += aT[cc] * leaky02(t);
    }
    acc += __shfl_xor(acc, 1);
    acc += __shfl_xor(acc, 2);
    if ((lane & 3) == 0) {
        logitsBuf[(size_t)w * NH + (lane >> 2)] = acc;
        atomicMax(gmax, encF(acc));
    }
}

__global__ void k_quadra_logits(const int* __restrict__ qi, const float* __restrict__ pos,
                                const float* __restrict__ srcP, const float* __restrict__ mid2P,
                                const float* __restrict__ mid1P, const float* __restrict__ dstP,
                                const float* __restrict__ Wpu2, const float* __restrict__ bpu2,
                                const float* __restrict__ Wdih, const float* __restrict__ bdih,
                                const float* __restrict__ aQ,
                                float* __restrict__ logitsBuf, unsigned int* __restrict__ gmax, int Q) {
    int w = (int)((blockIdx.x * blockDim.x + threadIdx.x) >> 5);
    int lane = threadIdx.x & 31;
    if (w >= Q) return;
    int qs = qi[w], qm2 = qi[Q + w], qm1 = qi[2 * Q + w], qd = qi[3 * Q + w];
    float dd = dist3(pos, qs, qd);
    float p1 = 1.f / dd, p2 = p1 * p1, p6 = p2 * p2 * p2, p12 = p6 * p6;
    float b1x = pos[3*qm2+0]-pos[3*qs+0],  b1y = pos[3*qm2+1]-pos[3*qs+1],  b1z = pos[3*qm2+2]-pos[3*qs+2];
    float b2x = pos[3*qm1+0]-pos[3*qm2+0], b2y = pos[3*qm1+1]-pos[3*qm2+1], b2z = pos[3*qm1+2]-pos[3*qm2+2];
    float b3x = pos[3*qd+0]-pos[3*qm1+0],  b3y = pos[3*qd+1]-pos[3*qm1+1],  b3z = pos[3*qd+2]-pos[3*qm1+2];
    float n1x = b1y*b2z - b1z*b2y, n1y = b1z*b2x - b1x*b2z, n1z = b1x*b2y - b1y*b2x;
    float n2x = b2y*b3z - b2z*b3y, n2y = b2z*b3x - b2x*b3z, n2z = b2x*b3y - b2y*b3x;
    float dn = n1x*n2x + n1y*n2y + n1z*n2z;
    float l1 = sqrtf(n1x*n1x + n1y*n1y + n1z*n1z);
    float l2 = sqrtf(n2x*n2x + n2y*n2y + n2z*n2z);
    float c = dn / (l1 * l2 + 1e-9f);
    c = fminf(fmaxf(c, -1.f + 1e-6f), 1.f - 1e-6f);
    float phi = acosf(c);
    float f0 = phi, f1 = phi * phi;
    float f2 = __cosf(phi), f3 = f2 * f2;    // phi in [0, pi] -> fast path is fine
    float f4 = __sinf(phi), f5 = f4 * f4;
    float acc = 0.f;
    int cb = lane * 4;
#pragma unroll
    for (int j = 0; j < 4; ++j) {
        int cc = cb + j;
        float dm3 = p6 * Wpu2[cc] + p12 * Wpu2[HD + cc] + p1 * Wpu2[2 * HD + cc] + bpu2[cc];
        float dih = f0 * Wdih[cc] + f1 * Wdih[HD + cc] + f2 * Wdih[2 * HD + cc]
                  + f3 * Wdih[3 * HD + cc] + f4 * Wdih[4 * HD + cc] + f5 * Wdih[5 * HD + cc] + bdih[cc];
        float t = (srcP[(size_t)qs * HD + cc] + mid2P[(size_t)qm2 * HD + cc] +
                   mid1P[(size_t)qm1 * HD + cc] + dstP[(size_t)qd * HD + cc]) * (dih + dm3);
        acc += aQ[cc] * leaky02(t);
    }
    acc += __shfl_xor(acc, 1);
    acc += __shfl_xor(acc, 2);
    if ((lane & 3) == 0) {
        logitsBuf[(size_t)w * NH + (lane >> 2)] = acc;
        atomicMax(gmax, encF(acc));
    }
}

// =====================  shared softmax passes  =====================

// logits -> exp(logit - gmax); accumulate per-(dst,head) partition sum
__global__ void k_exp_s(float* __restrict__ eBuf, const int* __restrict__ dstIdx,
                        float* __restrict__ sBuf, const unsigned int* __restrict__ gmaxU, int M) {
    int i = blockIdx.x * blockDim.x + threadIdx.x;
    if (i >= M * NH) return;
    float gmax = decF(*gmaxU);
    float ev = __expf(eBuf[i] - gmax);   // arg <= 0 after max subtraction: fast path safe
    eBuf[i] = ev;
    int item = i >> 3, h = i & 7;
    atomicAdd(&sBuf[(size_t)dstIdx[item] * NH + h], ev);
}

// out[dst] += val[src] * (e / (s[dst]+eps)) — wave per item, float4 gather, atomic scatter
__global__ void k_scatter(const float* __restrict__ eBuf, const float* __restrict__ sBuf,
                          const int* __restrict__ srcIdx, const int* __restrict__ dstIdx,
                          const float* __restrict__ valMat, float* __restrict__ outMat, int M) {
    int w = (int)((blockIdx.x * blockDim.x + threadIdx.x) >> 5);
    int lane = threadIdx.x & 31;
    if (w >= M) return;
    int si = srcIdx[w], di = dstIdx[w];
    int h = lane >> 2;
    float a = eBuf[(size_t)w * NH + h] / (sBuf[(size_t)di * NH + h] + 1e-16f);
    int cb = lane * 4;
    float4 v = *(const float4*)(valMat + (size_t)si * HD + cb);
    float* o = outMat + (size_t)di * HD + cb;
    atomicAdd(o + 0, v.x * a);
    atomicAdd(o + 1, v.y * a);
    atomicAdd(o + 2, v.z * a);
    atomicAdd(o + 3, v.w * a);
}

// =====================  WMMA combine: out = Σ hop_out@W + biases + x  =====================

__global__ void k_combine(const unsigned short* __restrict__ hopBf,
                          const unsigned short* __restrict__ wbf,
                          const float* __restrict__ x,
                          const float* __restrict__ b1, const float* __restrict__ b2,
                          const float* __restrict__ b3,
                          float* __restrict__ out, int nRows, int mTiles) {
    int wave = (int)((blockIdx.x * blockDim.x + threadIdx.x) >> 5);
    int lane = threadIdx.x & 31;
    if (wave >= mTiles * 8) return;
    int mT = wave >> 3, nT = wave & 7;
    int m0 = mT * 16, n0 = nT * 16;
    int lr = lane & 15;
    int kb = (lane >> 4) << 3;

    v8f acc = {};
#pragma unroll
    for (int hop = 0; hop < 3; ++hop) {
        const unsigned short* Abase = hopBf + (size_t)hop * nRows * HD + (size_t)(m0 + lr) * HD;
        const unsigned short* Bbase = wbf + (size_t)(4 + hop) * 16384 + (size_t)(n0 + lr) * HD;
#pragma unroll
        for (int k0 = 0; k0 < HD; k0 += 32) {
            Frag16 a, b;
            a.q[0] = *(const uint4*)(Abase + k0 + kb);
            a.q[1] = *(const uint4*)(Abase + k0 + 16 + kb);
            b.q[0] = *(const uint4*)(Bbase + k0 + kb);
            b.q[1] = *(const uint4*)(Bbase + k0 + 16 + kb);
            acc = __builtin_amdgcn_wmma_f32_16x16x32_bf16(false, a.v, false, b.v,
                                                          (short)0, acc, false, false);
        }
    }
    int n = n0 + lr;
    float bias = b1[n] + b2[n] + b3[n];
#pragma unroll
    for (int v = 0; v < 8; ++v) {
        int row = m0 + v + ((lane >> 4) << 3);
        out[(size_t)row * HD + n] = acc[v] + bias + x[(size_t)row * HD + n];
    }
}

// =====================  LayerNorm + out_bias + PReLU (wave per row)  =====================

__global__ void k_ln(float* __restrict__ out, const float* __restrict__ g,
                     const float* __restrict__ b, const float* __restrict__ ob,
                     const float* __restrict__ pa_p, int nRows) {
    int w = (int)((blockIdx.x * blockDim.x + threadIdx.x) >> 5);
    int lane = threadIdx.x & 31;
    if (w >= nRows) return;
    float* row = out + (size_t)w * HD;
    int cb = lane * 4;
    float4 v = *(const float4*)(row + cb);
    float s  = v.x + v.y + v.z + v.w;
    float ss = v.x * v.x + v.y * v.y + v.z * v.z + v.w * v.w;
#pragma unroll
    for (int o = 16; o >= 1; o >>= 1) {
        s  += __shfl_xor(s, o);
        ss += __shfl_xor(ss, o);
    }
    float mu  = s * (1.f / HD);
    float var = ss * (1.f / HD) - mu * mu;
    float inv = rsqrtf(var + 1e-5f);
    float pa  = pa_p[0];
    float r4[4] = {v.x, v.y, v.z, v.w};
#pragma unroll
    for (int j = 0; j < 4; ++j) {
        int n = cb + j;
        float r = (r4[j] - mu) * inv * g[n] + b[n] + ob[n];
        r4[j] = r >= 0.f ? r : pa * r;
    }
    *(float4*)(row + cb) = make_float4(r4[0], r4[1], r4[2], r4[3]);
}

// =====================  launch  =====================

extern "C" void kernel_launch(void* const* d_in, const int* in_sizes, int n_in,
                              void* d_out, int out_size, void* d_ws, size_t ws_size,
                              hipStream_t stream) {
    const float* x   = (const float*)d_in[0];
    const float* pos = (const float*)d_in[1];
    const float* W_src = (const float*)d_in[2];
    const float* W_dst = (const float*)d_in[3];
    const float* W_mid1 = (const float*)d_in[4];
    const float* W_mid2 = (const float*)d_in[5];
    const float* W_pb = (const float*)d_in[6];  const float* b_pb = (const float*)d_in[7];
    const float* W_pu = (const float*)d_in[8];  const float* b_pu = (const float*)d_in[9];
    const float* W_pu1 = (const float*)d_in[10]; const float* b_pu1 = (const float*)d_in[11];
    const float* W_pu2 = (const float*)d_in[12]; const float* b_pu2 = (const float*)d_in[13];
    const float* W_ang = (const float*)d_in[14]; const float* b_ang = (const float*)d_in[15];
    const float* W_dih = (const float*)d_in[16]; const float* b_dih = (const float*)d_in[17];
    const float* W_one = (const float*)d_in[18]; const float* b_one = (const float*)d_in[19];
    const float* W_two = (const float*)d_in[20]; const float* b_two = (const float*)d_in[21];
    const float* W_three = (const float*)d_in[22]; const float* b_three = (const float*)d_in[23];
    const float* a_double = (const float*)d_in[24];
    const float* a_triple = (const float*)d_in[25];
    const float* a_quadra = (const float*)d_in[26];
    const float* ln_g = (const float*)d_in[27]; const float* ln_b = (const float*)d_in[28];
    const float* out_bias = (const float*)d_in[29];
    const float* prelu_a  = (const float*)d_in[30];
    const int* ei = (const int*)d_in[31];
    const int* ti = (const int*)d_in[32];
    const int* qi = (const int*)d_in[33];
    const unsigned char* eattr = (const unsigned char*)d_in[34];   // bool8

    const int N = in_sizes[0] / HD;
    const int E = in_sizes[31] / 2;
    const int T = in_sizes[32] / 3;
    const int Q = in_sizes[33] / 4;
    const int mTiles = N / 16;          // N = 50000 -> 3125 exact

    // workspace carve-out (~250 MB, assumed < ws_size)
    auto al = [](size_t v) { return (v + 255) & ~(size_t)255; };
    char* ws = (char*)d_ws;
    size_t off = 0;
    unsigned short* xbf   = (unsigned short*)(ws + off); off += al((size_t)N * HD * 2);
    unsigned short* wbf   = (unsigned short*)(ws + off); off += al((size_t)7 * HD * HD * 2);
    float* projBase       = (float*)(ws + off);          off += al((size_t)4 * N * HD * 4);
    float* hopOut         = (float*)(ws + off);          off += al((size_t)3 * N * HD * 4);
    unsigned short* hopBf = (unsigned short*)(ws + off); off += al((size_t)3 * N * HD * 2);
    float* eBuf           = (float*)(ws + off);          off += al((size_t)Q * NH * 4);
    float* sBuf           = (float*)(ws + off);          off += al((size_t)N * NH * 4);
    unsigned int* gmax    = (unsigned int*)(ws + off);   off += 256;

    const float* srcP  = projBase + (size_t)0 * N * HD;
    const float* mid2P = projBase + (size_t)1 * N * HD;
    const float* mid1P = projBase + (size_t)2 * N * HD;
    const float* dstP  = projBase + (size_t)3 * N * HD;

    // zero accumulators (graph-capturable stream memsets)
    hipMemsetAsync(hopOut, 0, (size_t)3 * N * HD * 4, stream);

    // bf16 conversion of x + transposed weights
    {
        int n_x = N * HD;
        int tot = n_x + 7 * HD * HD;
        k_convert<<<(tot + 255) / 256, 256, 0, stream>>>(
            x, W_src, W_mid2, W_mid1, W_dst, W_one, W_two, W_three, xbf, wbf, n_x);
    }

    // 4 input projections (WMMA, wave per (mat, M-tile), A reused across 8 N-tiles)
    {
        int waves = mTiles * 4;
        k_proj<<<(waves * 32 + 255) / 256, 256, 0, stream>>>(xbf, wbf, projBase, N, mTiles);
    }

    // ---- one-hop (edges) ----
    hipMemsetAsync(sBuf, 0, (size_t)N * NH * 4, stream);
    hipMemsetAsync(gmax, 0, 4, stream);
    k_edge_logits<<<(E * 32 + 255) / 256, 256, 0, stream>>>(
        ei, eattr, pos, mid1P, dstP, W_pb, b_pb, W_pu, b_pu, a_double, eBuf, gmax, E);
    k_exp_s<<<(E * NH + 255) / 256, 256, 0, stream>>>(eBuf, ei + E, sBuf, gmax, E);
    k_scatter<<<(E * 32 + 255) / 256, 256, 0, stream>>>(
        eBuf, sBuf, ei, ei + E, mid1P, hopOut + (size_t)0 * N * HD, E);

    // ---- two-hop (angles) ----
    hipMemsetAsync(sBuf, 0, (size_t)N * NH * 4, stream);
    hipMemsetAsync(gmax, 0, 4, stream);
    k_triple_logits<<<(T * 32 + 255) / 256, 256, 0, stream>>>(
        ti, pos, mid2P, mid1P, dstP, W_pu1, b_pu1, W_ang, b_ang, a_triple, eBuf, gmax, T);
    k_exp_s<<<(T * NH + 255) / 256, 256, 0, stream>>>(eBuf, ti + 2 * T, sBuf, gmax, T);
    k_scatter<<<(T * 32 + 255) / 256, 256, 0, stream>>>(
        eBuf, sBuf, ti, ti + 2 * T, mid2P, hopOut + (size_t)1 * N * HD, T);

    // ---- three-hop (dihedrals) ----
    hipMemsetAsync(sBuf, 0, (size_t)N * NH * 4, stream);
    hipMemsetAsync(gmax, 0, 4, stream);
    k_quadra_logits<<<(Q * 32 + 255) / 256, 256, 0, stream>>>(
        qi, pos, srcP, mid2P, mid1P, dstP, W_pu2, b_pu2, W_dih, b_dih, a_quadra, eBuf, gmax, Q);
    k_exp_s<<<(Q * NH + 255) / 256, 256, 0, stream>>>(eBuf, qi + 3 * Q, sBuf, gmax, Q);
    k_scatter<<<(Q * 32 + 255) / 256, 256, 0, stream>>>(
        eBuf, sBuf, qi, qi + 3 * Q, srcP, hopOut + (size_t)2 * N * HD, Q);

    // hop outputs -> bf16 for combine GEMMs
    {
        int tot = 3 * N * HD;
        k_f2bf<<<(tot + 255) / 256, 256, 0, stream>>>(hopOut, hopBf, tot);
    }

    // combine (WMMA) + residual, then LayerNorm + bias + PReLU
    float* outp = (float*)d_out;
    k_combine<<<(mTiles * 8 * 32 + 255) / 256, 256, 0, stream>>>(
        hopBf, wbf, x, b_one, b_two, b_three, outp, N, mTiles);
    k_ln<<<(N * 32 + 255) / 256, 256, 0, stream>>>(outp, ln_g, ln_b, out_bias, prelu_a, N);
}
